// MultiHeadSelfAttention_21088289424024
// MI455X (gfx1250) — matrix operand order repaired and checked by reading, not compiled, not run on hardware
//
#include <hip/hip_runtime.h>
#include <hip/hip_bf16.h>

#define Bc 4
#define Sc 2048
#define Dc 1024
#define Hc 16
#define HDc 64
#define Mc (Bc * Sc)          // 8192 rows

typedef __attribute__((ext_vector_type(16))) __bf16 v16bf;
typedef __attribute__((ext_vector_type(8)))  float  v8f;

__device__ __forceinline__ unsigned short f32_to_bf16(float f) {
    unsigned int u = __float_as_uint(f);
    unsigned int r = (u + 0x7FFFu + ((u >> 16) & 1u)) >> 16;   // RNE
    return (unsigned short)r;
}

__device__ __forceinline__ v8f wmma_bf16(v16bf a, v16bf b, v8f c) {
    return __builtin_amdgcn_wmma_f32_16x16x32_bf16(
        /*neg_a=*/false, a, /*neg_b=*/false, b,
        /*c_mod=*/(short)0, c, /*reuse_a=*/false, /*reuse_b=*/false);
}

// A-matrix 16x32 bf16 fragment (ISA 7.12.2): lane l -> row r0+(l&15);
// 16B chunk at K = k0 + (l>>4)*8 into v0..v3, 16B chunk at K+16 into v4..v7.
__device__ __forceinline__ v16bf load_a_frag(const unsigned short* base, int stride,
                                             int r0, int k0, int lane) {
    int m = r0 + (lane & 15);
    int koff = (lane >> 4) * 8;
    const unsigned short* p = base + (size_t)m * stride + k0 + koff;
    union { uint4 u[2]; v16bf v; } x;
    x.u[0] = *(const uint4*)p;
    x.u[1] = *(const uint4*)(p + 16);
    return x.v;
}

// B-matrix 32x16 bf16 fragment: lane l -> col n0+(l&15);
// 16B chunk at K = k0 + (l>>4)*8 into v0..v3, 16B chunk at K+16 into v4..v7.
// `base` stores B transposed: base[n * stride + k].
__device__ __forceinline__ v16bf load_b_frag(const unsigned short* base, int stride,
                                             int n0, int k0, int lane) {
    int n = n0 + (lane & 15);
    int koff = (lane >> 4) * 8;
    const unsigned short* p = base + (size_t)n * stride + k0 + koff;
    union { uint4 u[2]; v16bf v; } x;
    x.u[0] = *(const uint4*)p;
    x.u[1] = *(const uint4*)(p + 16);
    return x.v;
}

// ---------------- conversion kernels ----------------
__global__ void k_f2bf(const float* __restrict__ src, unsigned short* __restrict__ dst, int n) {
    int i = blockIdx.x * blockDim.x + threadIdx.x;
    if (i < n) dst[i] = f32_to_bf16(src[i]);
}

// src [K,N] f32 row-major -> dst [N,K] bf16 (transposed)
__global__ void k_f2bf_T(const float* __restrict__ src, unsigned short* __restrict__ dst,
                         int K, int N) {
    int i = blockIdx.x * blockDim.x + threadIdx.x;
    if (i < K * N) {
        int k = i / N, n = i - k * N;
        dst[(size_t)n * K + k] = f32_to_bf16(src[i]);
    }
}

// ---------------- QKV GEMM: [8192,1024] x [1024,3072] -> scatter q/k/vT ----------------
__global__ __launch_bounds__(256) void k_qkv(const unsigned short* __restrict__ xb,
                                             const unsigned short* __restrict__ wqt, // [3072,1024]
                                             const float* __restrict__ bqkv,
                                             unsigned short* __restrict__ qb,
                                             unsigned short* __restrict__ kbuf,
                                             unsigned short* __restrict__ vtb) {
    int lane = threadIdx.x & 31;
    int wid  = blockIdx.x * 8 + (threadIdx.x >> 5);   // 6144 waves
    int mt = wid / 48, nt = wid - mt * 48;
    int m0 = mt * 64, n0 = nt * 64;
    int lo = lane & 15, hi = lane >> 4;

    v8f acc[4][4];
    #pragma unroll
    for (int i = 0; i < 4; ++i)
        #pragma unroll
        for (int j = 0; j < 4; ++j) acc[i][j] = {};

    for (int k0 = 0; k0 < Dc; k0 += 32) {
        if (k0 + 32 < Dc) {
            __builtin_prefetch(xb  + (size_t)(m0 + lo) * Dc + k0 + 32, 0, 3);
            __builtin_prefetch(wqt + (size_t)(n0 + lo) * Dc + k0 + 32, 0, 3);
        }
        v16bf a[4], bf[4];
        #pragma unroll
        for (int i = 0; i < 4; ++i) a[i]  = load_a_frag(xb,  Dc, m0 + 16 * i, k0, lane);
        #pragma unroll
        for (int j = 0; j < 4; ++j) bf[j] = load_b_frag(wqt, Dc, n0 + 16 * j, k0, lane);
        #pragma unroll
        for (int i = 0; i < 4; ++i)
            #pragma unroll
            for (int j = 0; j < 4; ++j) acc[i][j] = wmma_bf16(a[i], bf[j], acc[i][j]);
    }

    #pragma unroll
    for (int j = 0; j < 4; ++j) {
        int col  = n0 + 16 * j + lo;
        float bias = bqkv[col];
        int part = col >> 10;         // 0=q 1=k 2=v (uniform per tile: 1024 % 64 == 0)
        int cc   = col & 1023;
        int hh   = cc >> 6, d = cc & 63;
        #pragma unroll
        for (int i = 0; i < 4; ++i) {
            #pragma unroll
            for (int r = 0; r < 8; ++r) {
                int row = m0 + 16 * i + r + 8 * hi;
                int bb = row >> 11, ss = row & 2047;
                unsigned short bv = f32_to_bf16(acc[i][j][r] + bias);
                size_t bh = (size_t)(bb * Hc + hh);
                if (part == 0)       qb  [(bh * Sc + ss) * HDc + d] = bv;
                else if (part == 1)  kbuf[(bh * Sc + ss) * HDc + d] = bv;
                else                 vtb [(bh * HDc + d) * Sc + ss] = bv;
            }
        }
    }
}

// ---------------- Flash attention: one wave per (b,h,16-query tile) ----------------
__global__ __launch_bounds__(256) void k_attn(const unsigned short* __restrict__ qb,
                                              const unsigned short* __restrict__ kb,
                                              const unsigned short* __restrict__ vtb,
                                              unsigned short* __restrict__ attnb) {
    __shared__ unsigned short lds_p[8][16 * 32];
    int lane = threadIdx.x & 31;
    int w    = threadIdx.x >> 5;
    int wid  = blockIdx.x * 8 + w;                 // 8192 waves
    int b   = wid >> 11;
    int rem = wid & 2047;
    int h   = rem >> 7;
    int qt  = rem & 127;
    int q0  = qt * 16;
    int lo = lane & 15, hi = lane >> 4;

    const unsigned short* Qb = qb  + (size_t)(b * Hc + h) * Sc * HDc;
    const unsigned short* Kb = kb  + (size_t)(b * Hc + h) * Sc * HDc;
    const unsigned short* Vb = vtb + (size_t)(b * Hc + h) * HDc * Sc;

    v16bf qf0 = load_a_frag(Qb, HDc, q0, 0,  lane);
    v16bf qf1 = load_a_frag(Qb, HDc, q0, 32, lane);

    float mrow[8], lrow[8];
    v8f O[4];
    #pragma unroll
    for (int r = 0; r < 8; ++r) { mrow[r] = -3.0e38f; lrow[r] = 0.0f; }
    #pragma unroll
    for (int t = 0; t < 4; ++t) O[t] = {};

    const float scale = 0.125f;                    // 1/sqrt(64)
    int nchunks = ((q0 + 15) >> 5) + 1;

    for (int c = 0; c < nchunks; ++c) {
        int kc = c * 32;
        v16bf kA0 = load_b_frag(Kb, HDc, kc,      0,  lane);
        v16bf kA1 = load_b_frag(Kb, HDc, kc,      32, lane);
        v16bf kB0 = load_b_frag(Kb, HDc, kc + 16, 0,  lane);
        v16bf kB1 = load_b_frag(Kb, HDc, kc + 16, 32, lane);
        v8f s0 = {}; s0 = wmma_bf16(qf0, kA0, s0); s0 = wmma_bf16(qf1, kA1, s0);
        v8f s1 = {}; s1 = wmma_bf16(qf0, kB0, s1); s1 = wmma_bf16(qf1, kB1, s1);

        float p0[8], p1[8];
        #pragma unroll
        for (int r = 0; r < 8; ++r) {
            int qrow = q0 + r + 8 * hi;
            float v0 = (kc + lo      <= qrow) ? s0[r] * scale : -1.0e9f;
            float v1 = (kc + 16 + lo <= qrow) ? s1[r] * scale : -1.0e9f;
            float t = fmaxf(v0, v1);
            t = fmaxf(t, __shfl_xor(t, 1, 32));
            t = fmaxf(t, __shfl_xor(t, 2, 32));
            t = fmaxf(t, __shfl_xor(t, 4, 32));
            t = fmaxf(t, __shfl_xor(t, 8, 32));
            float mnew  = fmaxf(mrow[r], t);
            float alpha = __expf(mrow[r] - mnew);
            float e0 = __expf(v0 - mnew), e1 = __expf(v1 - mnew);
            float ssum = e0 + e1;
            ssum += __shfl_xor(ssum, 1, 32);
            ssum += __shfl_xor(ssum, 2, 32);
            ssum += __shfl_xor(ssum, 4, 32);
            ssum += __shfl_xor(ssum, 8, 32);
            lrow[r] = lrow[r] * alpha + ssum;
            mrow[r] = mnew;
            p0[r] = e0; p1[r] = e1;
            O[0][r] *= alpha; O[1][r] *= alpha; O[2][r] *= alpha; O[3][r] *= alpha;
        }

        // C-layout -> A-layout via wave-private LDS (same-wave DS ops are in order)
        unsigned short* myp = lds_p[w];
        #pragma unroll
        for (int r = 0; r < 8; ++r) {
            int row = r + 8 * hi;
            myp[row * 32 + lo]      = f32_to_bf16(p0[r]);
            myp[row * 32 + 16 + lo] = f32_to_bf16(p1[r]);
        }
        asm volatile("" ::: "memory");
        v16bf pf;
        {
            const unsigned short* p = myp + lo * 32 + hi * 8;
            union { uint4 u[2]; v16bf v; } x;
            x.u[0] = *(const uint4*)p;
            x.u[1] = *(const uint4*)(p + 16);
            pf = x.v;
        }

        #pragma unroll
        for (int t = 0; t < 4; ++t) {
            v16bf vf = load_b_frag(Vb, Sc, t * 16, kc, lane);
            O[t] = wmma_bf16(pf, vf, O[t]);
        }
    }

    #pragma unroll
    for (int r = 0; r < 8; ++r) {
        float inv = 1.0f / lrow[r];
        int srow = q0 + r + 8 * hi;
        size_t ob = ((size_t)b * Sc + srow) * Dc + h * HDc;
        #pragma unroll
        for (int t = 0; t < 4; ++t)
            attnb[ob + t * 16 + lo] = f32_to_bf16(O[t][r] * inv);
    }
}

// ---------------- Output GEMM: [8192,1024] x [1024,1024] + bias -> f32 ----------------
__global__ __launch_bounds__(256) void k_out(const unsigned short* __restrict__ attnb,
                                             const unsigned short* __restrict__ wot, // [1024,1024] T
                                             const float* __restrict__ bout,
                                             float* __restrict__ out) {
    int lane = threadIdx.x & 31;
    int wid  = blockIdx.x * 8 + (threadIdx.x >> 5);  // 2048 waves
    int mt = wid >> 4, nt = wid & 15;
    int m0 = mt * 64, n0 = nt * 64;
    int lo = lane & 15, hi = lane >> 4;

    v8f acc[4][4];
    #pragma unroll
    for (int i = 0; i < 4; ++i)
        #pragma unroll
        for (int j = 0; j < 4; ++j) acc[i][j] = {};

    for (int k0 = 0; k0 < Dc; k0 += 32) {
        if (k0 + 32 < Dc) {
            __builtin_prefetch(attnb + (size_t)(m0 + lo) * Dc + k0 + 32, 0, 3);
            __builtin_prefetch(wot   + (size_t)(n0 + lo) * Dc + k0 + 32, 0, 3);
        }
        v16bf a[4], bf[4];
        #pragma unroll
        for (int i = 0; i < 4; ++i) a[i]  = load_a_frag(attnb, Dc, m0 + 16 * i, k0, lane);
        #pragma unroll
        for (int j = 0; j < 4; ++j) bf[j] = load_b_frag(wot,   Dc, n0 + 16 * j, k0, lane);
        #pragma unroll
        for (int i = 0; i < 4; ++i)
            #pragma unroll
            for (int j = 0; j < 4; ++j) acc[i][j] = wmma_bf16(a[i], bf[j], acc[i][j]);
    }

    #pragma unroll
    for (int j = 0; j < 4; ++j) {
        int col = n0 + 16 * j + lo;
        float bias = bout[col];
        #pragma unroll
        for (int i = 0; i < 4; ++i)
            #pragma unroll
            for (int r = 0; r < 8; ++r) {
                int row = m0 + 16 * i + r + 8 * hi;
                out[(size_t)row * Dc + col] = acc[i][j][r] + bias;
            }
    }
}

extern "C" void kernel_launch(void* const* d_in, const int* in_sizes, int n_in,
                              void* d_out, int out_size, void* d_ws, size_t ws_size,
                              hipStream_t stream) {
    (void)in_sizes; (void)n_in; (void)out_size; (void)ws_size;
    const float* x    = (const float*)d_in[0];
    const float* Wq   = (const float*)d_in[1];
    const float* bq   = (const float*)d_in[2];
    const float* Wo   = (const float*)d_in[3];
    const float* bo   = (const float*)d_in[4];
    float* out = (float*)d_out;

    char* ws = (char*)d_ws;
    unsigned short* xb   = (unsigned short*)(ws);                       // 16 MB  [8192,1024]
    unsigned short* wqt  = (unsigned short*)(ws + 16777216);            // 6 MB   [3072,1024]
    unsigned short* wot  = (unsigned short*)(ws + 23068672);            // 2 MB   [1024,1024]
    unsigned short* qb   = (unsigned short*)(ws + 25165824);            // 16 MB  [B,H,S,HD]
    unsigned short* kbuf = (unsigned short*)(ws + 41943040);            // 16 MB  [B,H,S,HD]
    unsigned short* vtb  = (unsigned short*)(ws + 58720256);            // 16 MB  [B,H,HD,S]
    unsigned short* attn = (unsigned short*)(ws + 75497472);            // 16 MB  [8192,1024]

    k_f2bf  <<<(Mc * Dc + 255) / 256, 256, 0, stream>>>(x, xb, Mc * Dc);
    k_f2bf_T<<<(Dc * 3 * Dc + 255) / 256, 256, 0, stream>>>(Wq, wqt, Dc, 3 * Dc);
    k_f2bf_T<<<(Dc * Dc + 255) / 256, 256, 0, stream>>>(Wo, wot, Dc, Dc);

    k_qkv <<<768,  256, 0, stream>>>(xb, wqt, bq, qb, kbuf, vtb);   // 6144 waves, 64x64 tiles
    k_attn<<<1024, 256, 0, stream>>>(qb, kbuf, vtb, attn);          // 8192 waves
    k_out <<<256,  256, 0, stream>>>(attn, wot, bo, out);           // 2048 waves
}
